// BiLinearAttention_82712480186621
// MI455X (gfx1250) — matrix operand order repaired and checked by reading, not compiled, not run on hardware
//
#include <hip/hip_runtime.h>
#include <hip/hip_bf16.h>

// ---------------------------------------------------------------------------
// BiLinearAttention for MI455X (gfx1250, wave32, WMMA)
//   score = softmax_mask( (Q@W) @ K^T );  out = score @ V;  returns (out, score)
// 3x bf16 WMMA GEMMs (f32 accumulate) in canonical NT form + f32 masked row
// softmax. f32 operands are converted to bf16 on the fly while staging into
// double-buffered LDS tiles; bf16 operands use async global->LDS copies
// (GLOBAL_LOAD_ASYNC_TO_LDS_B128, ASYNCcnt) when the toolchain exposes them.
// ---------------------------------------------------------------------------

typedef unsigned short u16;
typedef __attribute__((ext_vector_type(16))) __bf16 v16bf;
typedef __attribute__((ext_vector_type(8)))  float  v8f;
typedef __attribute__((ext_vector_type(4)))  int    v4i;

// ---- f32 -> bf16 (RNE), hardware packed convert when available ------------
__device__ __forceinline__ u16 f2bf(float f) {
    unsigned int u = __float_as_uint(f);
    unsigned int r = u + 0x7FFFu + ((u >> 16) & 1u);
    return (u16)(r >> 16);
}

#if __has_builtin(__builtin_amdgcn_cvt_pk_bf16_f32)
__device__ __forceinline__ unsigned pack2bf(float a, float b) {   // v_cvt_pk_bf16_f32
    typedef __bf16 bf2 __attribute__((ext_vector_type(2)));
    union { bf2 v; unsigned u; } c;
    c.v = __builtin_amdgcn_cvt_pk_bf16_f32(a, b);
    return c.u;
}
#else
__device__ __forceinline__ unsigned pack2bf(float a, float b) {
    return (unsigned)f2bf(a) | ((unsigned)f2bf(b) << 16);
}
#endif

// ---- async global->LDS (16B per lane), ASYNCcnt-tracked -------------------
#if __has_builtin(__builtin_amdgcn_global_load_async_to_lds_b128)
#define HAVE_ASYNC_LDS 1
typedef __attribute__((address_space(1))) v4i g_v4i;   // global int4
typedef __attribute__((address_space(3))) v4i l_v4i;   // LDS int4
__device__ __forceinline__ void async_cp16(const void* g, void* l) {
    __builtin_amdgcn_global_load_async_to_lds_b128(
        (g_v4i*)g, (l_v4i*)l, /*offset=*/0, /*cpol=*/0);
}
#else
#define HAVE_ASYNC_LDS 0
#endif

__device__ __forceinline__ void wait_async0() {
#if HAVE_ASYNC_LDS
#if __has_builtin(__builtin_amdgcn_s_wait_asynccnt)
    __builtin_amdgcn_s_wait_asynccnt(0);
#else
    asm volatile("s_wait_asynccnt 0x0" ::: "memory");
#endif
#endif
}

// ------------------------- transpose + f32->bf16 ---------------------------
// out[c][r] = bf16(in[r][c]), batched (grid.z)
__global__ __launch_bounds__(256)
void transpose_cvt(const float* __restrict__ in, u16* __restrict__ out,
                   int R, int C, long long inStride, long long outStride) {
    __shared__ float tile[32][33];
    const long long z = blockIdx.z;
    in  += z * inStride;
    out += z * outStride;
    const int r0 = blockIdx.y * 32, c0 = blockIdx.x * 32;
    const int tx = threadIdx.x, ty = threadIdx.y;           // block (32,8)
    for (int i = ty; i < 32; i += 8)
        tile[i][tx] = in[(size_t)(r0 + i) * C + (c0 + tx)];
    __syncthreads();
    for (int i = ty; i < 32; i += 8)
        out[(size_t)(c0 + i) * R + (r0 + tx)] = f2bf(tile[tx][i]);
}

// ------------------------- WMMA bf16 NT GEMM -------------------------------
// C[m,n] = sum_k A[m,k] * B[n,k]   (both operands contiguous along k)
// Workgroup tile 128x128, 8 waves of wave32, each wave: 2x4 16x16 f32 accs.
// LDS: double-buffered 128x32 bf16 tiles, padded row stride 40 (80B, 16B-aligned).
constexpr int TP = 40;   // padded LDS row stride in u16 elements

// f32 source: load f32, convert to bf16, store to LDS through VGPRs.
__device__ __forceinline__ void load_tile32_cvt(u16* __restrict__ dst,
                                                const void* __restrict__ srcBase,
                                                int ld, int row0, int k0, int tid) {
    for (int i = tid; i < 512; i += 256) {
        const int r = i >> 2;                 // 4 groups of 8 per row
        const int c = (i & 3) << 3;           // 0,8,16,24
        const float* s = (const float*)srcBase + (size_t)(row0 + r) * ld + k0 + c;
        float4 f0 = *(const float4*)(s);
        float4 f1 = *(const float4*)(s + 4);
        __builtin_prefetch(s + 64, 0, 1);     // speculative, two K-slabs ahead
        uint4 p;
        p.x = pack2bf(f0.x, f0.y);
        p.y = pack2bf(f0.z, f0.w);
        p.z = pack2bf(f1.x, f1.y);
        p.w = pack2bf(f1.z, f1.w);
        *(uint4*)(dst + r * TP + c) = p;      // ds_store_b128
    }
}

// bf16 source: raw 16B copies, async to LDS when available.
__device__ __forceinline__ void load_tile32_b16(u16* __restrict__ dst,
                                                const void* __restrict__ srcBase,
                                                int ld, int row0, int k0, int tid) {
    for (int i = tid; i < 512; i += 256) {
        const int r = i >> 2;
        const int c = (i & 3) << 3;
        const u16* s = (const u16*)srcBase + (size_t)(row0 + r) * ld + k0 + c;
        u16* d = dst + r * TP + c;
#if HAVE_ASYNC_LDS
        async_cp16(s, d);                     // global_load_async_to_lds_b128
#else
        *(uint4*)d = *(const uint4*)s;
#endif
    }
}

// A fragment, 16x32 bf16 (ISA 7.12.2): lanes 0-15 row M=lane, K={0..7,16..23};
// lanes 16-31 row M=lane-16, K={8..15,24..31}. Two ds_load_b128 per fragment.
__device__ __forceinline__ v16bf load_fragA(const u16* tile, int rowBase, int lane) {
    const int h = lane >> 4, m = lane & 15;
    const u16* p = tile + (rowBase + m) * TP + 8 * h;
    union { uint4 u[2]; v16bf v; } f;
    f.u[0] = *(const uint4*)(p);              // K = 8h .. 8h+7
    f.u[1] = *(const uint4*)(p + 16);         // K = 16+8h .. 16+8h+7
    return f.v;
}

// B fragment, 32x16 bf16: lanes 0-15 col N=lane hold K={0..7,16..23},
// lanes 16-31 col N=lane-16 hold K={8..15,24..31} (same order as A).
__device__ __forceinline__ v16bf load_fragB(const u16* tile, int colBase, int lane) {
    const int h = lane >> 4, n = lane & 15;
    const u16* p = tile + (colBase + n) * TP + 8 * h;
    union { uint4 u[2]; v16bf v; } f;
    f.u[0] = *(const uint4*)(p);
    f.u[1] = *(const uint4*)(p + 16);
    return f.v;
}

template<bool AF32, bool BF32, bool CBF16>
__global__ __launch_bounds__(256)
void gemm_nt(const void* __restrict__ Ap, const void* __restrict__ Bp,
             void* __restrict__ Cp, int Kdim, int lda, int ldb, int ldc,
             long long sA, long long sB, long long sC) {
    __shared__ __align__(16) u16 As[2][128 * TP];
    __shared__ __align__(16) u16 Bs[2][128 * TP];

    const int tid  = threadIdx.x;
    const int lane = tid & 31;
    const int wave = tid >> 5;
    const int wm   = wave & 3;                 // 4 waves along M (32 rows each)
    const int wn   = wave >> 2;                // 2 waves along N (64 cols each)
    const long long z = blockIdx.z;

    const char* A = (const char*)Ap + (AF32 ? 4LL : 2LL) * (z * sA);
    const char* B = (const char*)Bp + (BF32 ? 4LL : 2LL) * (z * sB);
    char*       C = (char*)Cp       + (CBF16 ? 2LL : 4LL) * (z * sC);

    const int m0 = blockIdx.y * 128;
    const int n0 = blockIdx.x * 128;

    v8f acc[2][4];
    for (int i = 0; i < 2; ++i)
        for (int j = 0; j < 4; ++j) acc[i][j] = {};

    if constexpr (AF32) load_tile32_cvt(As[0], A, lda, m0, 0, tid);
    else                load_tile32_b16(As[0], A, lda, m0, 0, tid);
    if constexpr (BF32) load_tile32_cvt(Bs[0], B, ldb, n0, 0, tid);
    else                load_tile32_b16(Bs[0], B, ldb, n0, 0, tid);
    wait_async0();
    __syncthreads();

    const int nk = Kdim >> 5;
    for (int kt = 0; kt < nk; ++kt) {
        const int cur = kt & 1;
        if (kt + 1 < nk) {                     // prefetch next K-slab into other buffer
            const int k1 = (kt + 1) << 5;
            if constexpr (AF32) load_tile32_cvt(As[cur ^ 1], A, lda, m0, k1, tid);
            else                load_tile32_b16(As[cur ^ 1], A, lda, m0, k1, tid);
            if constexpr (BF32) load_tile32_cvt(Bs[cur ^ 1], B, ldb, n0, k1, tid);
            else                load_tile32_b16(Bs[cur ^ 1], B, ldb, n0, k1, tid);
        }
        v16bf af[2], fb[4];
        af[0] = load_fragA(As[cur], wm * 32 + 0,  lane);
        af[1] = load_fragA(As[cur], wm * 32 + 16, lane);
        #pragma unroll
        for (int j = 0; j < 4; ++j) fb[j] = load_fragB(Bs[cur], wn * 64 + 16 * j, lane);
        #pragma unroll
        for (int i = 0; i < 2; ++i)
            #pragma unroll
            for (int j = 0; j < 4; ++j)
                acc[i][j] = __builtin_amdgcn_wmma_f32_16x16x32_bf16(
                    /*neg_a=*/false, af[i], /*neg_b=*/false, fb[j],
                    /*c_mod=*/(short)0, acc[i][j],
                    /*reuse_a=*/false, /*reuse_b=*/false);
        wait_async0();                         // async LDS writes done before barrier
        __syncthreads();
    }

    // C/D layout: lane = {h=lane>>4, n=lane&15}; VGPR r holds M = r + 8h.
    const int h = lane >> 4, nl = lane & 15;
    for (int i = 0; i < 2; ++i)
        for (int j = 0; j < 4; ++j) {
            const int mBase = m0 + wm * 32 + 16 * i + 8 * h;
            const int col   = n0 + wn * 64 + 16 * j + nl;
            #pragma unroll
            for (int r = 0; r < 8; ++r) {
                const float v = acc[i][j][r];
                if (CBF16) ((u16*)C)[(size_t)(mBase + r) * ldc + col] = f2bf(v);
                else       ((float*)C)[(size_t)(mBase + r) * ldc + col] = v;
            }
        }
}

// ------------------------- masked row softmax ------------------------------
// One 256-thread block per (batch, query) row of S=2048 scores, in place.
__global__ __launch_bounds__(256)
void softmax_mask(float* __restrict__ score, const unsigned char* __restrict__ mask, int S) {
    __shared__ float red[256];
    const int tid = threadIdx.x;
    float* p = score + ((size_t)blockIdx.y * gridDim.x + blockIdx.x) * S;
    const unsigned char* m = mask + (size_t)blockIdx.y * S;

    float vals[8];
    float mx = -3.4e38f;
    #pragma unroll
    for (int i = 0; i < 8; ++i) {
        const int s = tid + i * 256;
        float v = p[s];
        v = m[s] ? v : -3.4e38f;
        vals[i] = v;
        mx = fmaxf(mx, v);
    }
    red[tid] = mx; __syncthreads();
    for (int off = 128; off > 0; off >>= 1) {
        if (tid < off) red[tid] = fmaxf(red[tid], red[tid + off]);
        __syncthreads();
    }
    mx = red[0]; __syncthreads();

    float sum = 0.f;
    #pragma unroll
    for (int i = 0; i < 8; ++i) {
        const float e = __expf(vals[i] - mx);
        vals[i] = e;
        sum += e;
    }
    red[tid] = sum; __syncthreads();
    for (int off = 128; off > 0; off >>= 1) {
        if (tid < off) red[tid] += red[tid + off];
        __syncthreads();
    }
    const float inv = 1.0f / red[0];
    #pragma unroll
    for (int i = 0; i < 8; ++i) p[tid + i * 256] = vals[i] * inv;
}

// ---------------------------------------------------------------------------
extern "C" void kernel_launch(void* const* d_in, const int* in_sizes, int n_in,
                              void* d_out, int out_size, void* d_ws, size_t ws_size,
                              hipStream_t stream) {
    const int N = 4, L = 2048, S = 2048, D = 1024;   // n, l, s, d (dq=dk=dv=D)

    const float* Q = (const float*)d_in[0];          // [N,L,D]
    const float* K = (const float*)d_in[1];          // [N,S,D]
    const float* V = (const float*)d_in[2];          // [N,S,D]
    const float* W = (const float*)d_in[3];          // [D,D]
    const unsigned char* mask = (const unsigned char*)d_in[4]; // [N,S] bool

    float* outVal = (float*)d_out;                       // [N,L,D]
    float* score  = outVal + (size_t)N * L * D;          // [N,L,S]

    // Workspace: Wt (DxD bf16) | QWb (N*L x D bf16) | Vt (N x D x S bf16) = 34 MiB
    u16* Wt  = (u16*)d_ws;
    u16* QWb = Wt  + (size_t)D * D;
    u16* Vtb = QWb + (size_t)N * L * D;

    // 1) Wt[n,k] = W[k,n] ; Vt[b,d,s] = V[b,s,d]  (f32 -> bf16)
    transpose_cvt<<<dim3(D / 32, D / 32, 1), dim3(32, 8), 0, stream>>>(
        W, Wt, D, D, 0, 0);
    transpose_cvt<<<dim3(D / 32, S / 32, N), dim3(32, 8), 0, stream>>>(
        V, Vtb, S, D, (long long)S * D, (long long)D * S);

    // 2) QWb = Q @ W  (A=f32 Q on-the-fly, B=Wt bf16, C=bf16), M = N*L flattened
    gemm_nt<true, false, true><<<dim3(D / 128, (N * L) / 128, 1), 256, 0, stream>>>(
        Q, Wt, QWb, D, D, D, D, 0, 0, 0);

    // 3) score = QW @ K^T  (A=QWb bf16, B=K f32 on-the-fly, C=f32), batched
    gemm_nt<false, true, false><<<dim3(S / 128, L / 128, N), 256, 0, stream>>>(
        QWb, K, score, D, D, D, S,
        (long long)L * D, (long long)S * D, (long long)L * S);

    // 4) masked softmax over s, in place in d_out score region
    softmax_mask<<<dim3(L, N), 256, 0, stream>>>(score, mask, S);

    // 5) out = P @ V  (A=P f32 on-the-fly, B=Vt bf16, C=f32), batched
    gemm_nt<true, false, false><<<dim3(D / 128, L / 128, N), 256, 0, stream>>>(
        score, Vtb, outVal, S, S, S, D,
        (long long)L * S, (long long)D * S, (long long)L * D);
}
